// WTS_36447092474139
// MI455X (gfx1250) — matrix operand order repaired and checked by reading, not compiled, not run on hardware
//
#include <hip/hip_runtime.h>
#include <hip/hip_bf16.h>
#include <math.h>

// ---------------------------------------------------------------------------
// DDSP-style synth for MI455X (gfx1250, wave32).
// Dense layers -> v_wmma_f32_16x16x32_f16, branch-free vectorized fragment
// loads (b128), EDGE-templated GEMM (selects only for ragged N).
// GRUs: persistent workgroup, f16 h in LDS for WMMA (ds_load_b128),
// f32 carry in registers, f16 pre-converted Whh.
// ---------------------------------------------------------------------------

typedef __attribute__((ext_vector_type(16))) _Float16 v16h;
typedef __attribute__((ext_vector_type(8)))  _Float16 h8;
typedef __attribute__((ext_vector_type(8)))  float    v8f;

#define BATCH 32
#define TFR   400          // FRAMES
#define HID   512
#define GDIM  1536         // 3*HID
#define BLOCK 160
#define ALEN  64000        // TFR*BLOCK
#define NBANDS 65
#define NWT   10
#define WTLEN 512
#define SRATE 16000

__device__ __forceinline__ float sigmoidf_(float x) { return 1.0f / (1.0f + __expf(-x)); }

// K-offset inside a 16x32 f16 A-fragment (ISA 7.12.2): i<8 -> laneHi*8+i,
// i>=8 -> 16+laneHi*8+(i-8).
__device__ __forceinline__ int a_koff(int i, int laneHi) {
    return (i < 8) ? (laneHi * 8 + i) : (16 + laneHi * 8 + (i - 8));
}

#define SHUF16(lo, hi) __builtin_shufflevector((lo), (hi), 0,1,2,3,4,5,6,7,8,9,10,11,12,13,14,15)

// ---------------------------------------------------------------------------
// Generic GEMM: C[M,N] = A[M,K] * W[N,K]^T + bias.  One 16x16 tile per wave.
// Preconditions: M % 16 == 0; if K >= 32 then lda % 4 == 0 and ldw % 4 == 0
// (true for every call site).  EDGE=true handles ragged N via clamp+select.
// ---------------------------------------------------------------------------
template <bool EDGE>
__global__ __launch_bounds__(128)
void k_gemm_wmma(const float* __restrict__ A, int lda,
                 const float* __restrict__ W, int ldw,
                 const float* __restrict__ bias,
                 float* __restrict__ C, int ldc,
                 int M, int N, int K)
{
    const int lane   = threadIdx.x & 31;
    const int wid    = threadIdx.x >> 5;
    const int laneLo = lane & 15;
    const int laneHi = lane >> 4;
    const int tilesN = (N + 15) >> 4;
    const int tilesM = M >> 4;
    int tile = blockIdx.x * 4 + wid;
    if (tile >= tilesM * tilesN) return;
    const int tm = tile / tilesN, tn = tile % tilesN;
    const int row0 = tm << 4, col0 = tn << 4;
    const int r = row0 + laneLo;                    // < M always (M % 16 == 0)
    const int n = col0 + laneLo;
    const bool inN = EDGE ? (n < N) : true;
    const int  nc  = EDGE ? (inN ? n : (N - 1)) : n;  // clamped -> loads legal

    const float* __restrict__ Ar = A + (size_t)r * lda;
    const float* __restrict__ Br = W + (size_t)nc * ldw;

    v8f acc = {};
    const int K32 = K & ~31;

    for (int k0 = 0; k0 < K32; k0 += 32) {
        // A: two contiguous 8-float runs per lane (unconditional b128 loads)
        float av[16];
        *(float4*)&av[0]  = *(const float4*)(Ar + k0 + laneHi * 8);
        *(float4*)&av[4]  = *(const float4*)(Ar + k0 + laneHi * 8 + 4);
        *(float4*)&av[8]  = *(const float4*)(Ar + k0 + 16 + laneHi * 8);
        *(float4*)&av[12] = *(const float4*)(Ar + k0 + 16 + laneHi * 8 + 4);
        // B: two contiguous 8-float runs per lane (same k order as A)
        float bv[16];
        *(float4*)&bv[0]  = *(const float4*)(Br + k0 + laneHi * 8);
        *(float4*)&bv[4]  = *(const float4*)(Br + k0 + laneHi * 8 + 4);
        *(float4*)&bv[8]  = *(const float4*)(Br + k0 + 16 + laneHi * 8);
        *(float4*)&bv[12] = *(const float4*)(Br + k0 + 16 + laneHi * 8 + 4);
        v16h a, b;
#pragma unroll
        for (int i = 0; i < 16; ++i) {
            a[i] = (_Float16)av[i];
            b[i] = EDGE ? (inN ? (_Float16)bv[i] : (_Float16)0.f) : (_Float16)bv[i];
        }
        acc = __builtin_amdgcn_wmma_f32_16x16x32_f16(false, a, false, b, (short)0, acc,
                                                     false, false);
    }

    if (K32 < K) {   // ragged-K tail: clamped scalar loads + selects (no branches)
        v16h a, b;
#pragma unroll
        for (int i = 0; i < 16; ++i) {
            int ka  = K32 + a_koff(i, laneHi);
            bool ia = (ka < K);
            float avv = Ar[ia ? ka : (K - 1)];
            a[i] = ia ? (_Float16)avv : (_Float16)0.f;
            int kb  = K32 + a_koff(i, laneHi);
            bool ib = (kb < K);
            float bvv = Br[ib ? kb : (K - 1)];
            b[i] = (ib && inN) ? (_Float16)bvv : (_Float16)0.f;
        }
        acc = __builtin_amdgcn_wmma_f32_16x16x32_f16(false, a, false, b, (short)0, acc,
                                                     false, false);
    }

#pragma unroll
    for (int v = 0; v < 8; ++v) {
        int row = row0 + v + laneHi * 8;            // < M always
        if (inN) C[(size_t)row * ldc + n] = acc[v] + (bias ? bias[n] : 0.f);
    }
}

// ---------------------------------------------------------------------------
// Persistent GRU: one workgroup (32 waves).  h kept as f16 in LDS (WMMA A
// operand, ds_load_b128) + f32 carry in per-thread registers.  Whh in f16.
// Per step: gh = h @ Whh^T (32x1536) via WMMA, then fused gates.
// ---------------------------------------------------------------------------
__global__ __launch_bounds__(1024)
void k_gru_seq(const float* __restrict__ xs,        // (B*T, 1536) = x@Wih^T+bih
               const _Float16* __restrict__ Whh16,  // (1536, 512) pre-converted
               const float* __restrict__ bhh,       // (1536)
               float* __restrict__ gh,              // scratch 32*1536
               float* __restrict__ hseq, int ldh,
               int T)
{
    __shared__ _Float16 hs16[BATCH * HID];          // 32 KB
    const int tid    = threadIdx.x;
    const int lane   = tid & 31;
    const int wid    = tid >> 5;
    const int laneLo = lane & 15;
    const int laneHi = lane >> 4;

    float hreg[16];                                 // f32 recurrence carry
#pragma unroll
    for (int e = 0; e < 16; ++e) hreg[e] = 0.f;
    for (int i = tid; i < BATCH * HID; i += 1024) hs16[i] = (_Float16)0.f;
    __syncthreads();

    for (int t = 0; t < T; ++t) {
        // --- gh = h(32x512) @ Whh^T -> 32x1536; 192 tiles, 6 per wave ---
        for (int tt = 0; tt < 6; ++tt) {
            int tile = wid * 6 + tt;
            int tm = tile / 96, tn = tile % 96;
            int row0 = tm << 4, col0 = tn << 4;
            const _Float16* __restrict__ Ar = &hs16[(row0 + laneLo) * HID];
            const _Float16* __restrict__ Br = Whh16 + (size_t)(col0 + laneLo) * HID;
            v8f acc = {};
            for (int k0 = 0; k0 < HID; k0 += 32) {
                h8 alo = *(const h8*)(Ar + k0 + laneHi * 8);        // ds_load_b128
                h8 ahi = *(const h8*)(Ar + k0 + 16 + laneHi * 8);
                h8 blo = *(const h8*)(Br + k0 + laneHi * 8);        // global b128
                h8 bhi = *(const h8*)(Br + k0 + 16 + laneHi * 8);
                v16h a = SHUF16(alo, ahi);
                v16h b = SHUF16(blo, bhi);
                acc = __builtin_amdgcn_wmma_f32_16x16x32_f16(false, a, false, b, (short)0,
                                                             acc, false, false);
            }
#pragma unroll
            for (int v = 0; v < 8; ++v) {
                int row = row0 + v + laneHi * 8;
                int col = col0 + laneLo;
                gh[row * GDIM + col] = acc[v] + bhh[col];
            }
        }
        __threadfence();
        __syncthreads();

        // --- gates: 32*512 elements; element (b,c) owned by a fixed thread ---
#pragma unroll
        for (int e = 0; e < 16; ++e) {
            int idx = e * 1024 + tid;            // idx = b*512 + c
            int b = idx >> 9, c = idx & 511;
            size_t xrow = ((size_t)b * T + t) * (size_t)GDIM;
            float xr = xs[xrow + c], xz = xs[xrow + 512 + c], xn = xs[xrow + 1024 + c];
            float hr = gh[b * GDIM + c], hz = gh[b * GDIM + 512 + c],
                  hn = gh[b * GDIM + 1024 + c];
            float rg = sigmoidf_(xr + hr);
            float zg = sigmoidf_(xz + hz);
            float ng = tanhf(xn + rg * hn);
            float hnew = (1.f - zg) * ng + zg * hreg[e];
            hreg[e] = hnew;
            hs16[idx] = (_Float16)hnew;
            hseq[((size_t)b * T + t) * (size_t)ldh + c] = hnew;
        }
        __syncthreads();
    }
}

// f32 -> f16 weight pre-conversion
__global__ void k_f2h(const float* __restrict__ src, _Float16* __restrict__ dst, int n)
{
    int i = blockIdx.x * blockDim.x + threadIdx.x;
    if (i < n) dst[i] = (_Float16)src[i];
}

// ---------------------------------------------------------------------------
// LayerNorm (over last dim) + LeakyReLU(0.01), in place.  One block per row.
// ---------------------------------------------------------------------------
__global__ __launch_bounds__(256)
void k_ln_leaky(float* __restrict__ X, int ld, int N,
                const float* __restrict__ g, const float* __restrict__ be)
{
    __shared__ float rs[256], rq[256];
    float* x = X + (size_t)blockIdx.x * ld;
    float s = 0.f, q = 0.f;
    for (int i = threadIdx.x; i < N; i += 256) { float v = x[i]; s += v; q += v * v; }
    rs[threadIdx.x] = s; rq[threadIdx.x] = q;
    __syncthreads();
    for (int off = 128; off > 0; off >>= 1) {
        if (threadIdx.x < off) {
            rs[threadIdx.x] += rs[threadIdx.x + off];
            rq[threadIdx.x] += rq[threadIdx.x + off];
        }
        __syncthreads();
    }
    float mean = rs[0] / N;
    float var  = rq[0] / N - mean * mean;
    float rstd = rsqrtf(var + 1e-5f);
    for (int i = threadIdx.x; i < N; i += 256) {
        float v = (x[i] - mean) * rstd * g[i] + be[i];
        x[i] = v > 0.f ? v : 0.01f * v;
    }
}

// mfcc (B,20,T) -> transpose + layernorm over 20 -> xa (B*T, 20)
__global__ void k_mfcc_ln(const float* __restrict__ mfcc, const float* __restrict__ g,
                          const float* __restrict__ be, float* __restrict__ xa)
{
    int idx = blockIdx.x * blockDim.x + threadIdx.x;
    if (idx >= BATCH * TFR) return;
    int b = idx / TFR, t = idx % TFR;
    float v[20]; float s = 0.f;
#pragma unroll
    for (int c = 0; c < 20; ++c) { v[c] = mfcc[((size_t)b * 20 + c) * TFR + t]; s += v[c]; }
    float mean = s * (1.f / 20.f), q = 0.f;
#pragma unroll
    for (int c = 0; c < 20; ++c) { float d = v[c] - mean; q += d * d; }
    float rstd = rsqrtf(q * (1.f / 20.f) + 1e-5f);
#pragma unroll
    for (int c = 0; c < 20; ++c) xa[(size_t)idx * 20 + c] = (v[c] - mean) * rstd * g[c] + be[c];
}

// total_amp_2 = sigmoid(loudness*lw + lb)
__global__ void k_ta2(const float* __restrict__ loud, const float* __restrict__ lw,
                      const float* __restrict__ lb, float* __restrict__ out, int n)
{
    int i = blockIdx.x * blockDim.x + threadIdx.x;
    if (i < n) out[i] = sigmoidf_(loud[i] * lw[0] + lb[0]);
}

// scale_function(x - 5) in place
__global__ void k_scale(float* __restrict__ x, int n)
{
    int i = blockIdx.x * blockDim.x + threadIdx.x;
    if (i < n) {
        float s = sigmoidf_(x[i] - 5.f);
        x[i] = 2.f * __powf(s, 2.3025850929940457f) + 1e-7f;
    }
}

// irfft(65 real bands)->128, roll+hann+pad-to-160+roll(-64), fused.
__global__ __launch_bounds__(BLOCK)
void k_impulse(const float* __restrict__ amp, float* __restrict__ imp)
{
    __shared__ float sa[NBANDS];
    int row = blockIdx.x, j = threadIdx.x;
    if (j < NBANDS) sa[j] = amp[(size_t)row * NBANDS + j];
    __syncthreads();
    int p = (j + 64) % BLOCK;
    float val = 0.f;
    if (p < 128) {
        int kk = (p + 64) & 127;
        float acc = sa[0];
        for (int k = 1; k < 64; ++k)
            acc += 2.f * sa[k] * __cosf(6.2831853071795864f * (float)(k * kk) / 128.f);
        acc += sa[64] * ((kk & 1) ? -1.f : 1.f);
        acc *= (1.f / 128.f);
        float win = 0.5f - 0.5f * __cosf(6.2831853071795864f * (float)p / 128.f);
        val = acc * win;
    }
    imp[(size_t)row * BLOCK + j] = val;
}

// per-frame causal conv of (noise*2-1) with impulse; writes filtered noise into SIG
__global__ __launch_bounds__(BLOCK)
void k_noiseconv(const float* __restrict__ noise_unit, const float* __restrict__ imp,
                 float* __restrict__ sig)
{
    __shared__ float sn[BLOCK], si[BLOCK];
    int row = blockIdx.x, j = threadIdx.x;
    sn[j] = noise_unit[(size_t)row * BLOCK + j] * 2.f - 1.f;
    si[j] = imp[(size_t)row * BLOCK + j];
    __syncthreads();
    float acc = 0.f;
    for (int m = 0; m <= j; ++m) acc += sn[m] * si[j - m];
    int b = row / TFR, t = row % TFR;
    sig[(size_t)b * ALEN + t * BLOCK + j] = acc;
}

// tanh(wavetables) and softmax(attention)
__global__ __launch_bounds__(WTLEN)
void k_wtprep(const float* __restrict__ wts, const float* __restrict__ attn,
              float* __restrict__ wtt, float* __restrict__ ww)
{
    int i = threadIdx.x;
    for (int w = 0; w < NWT; ++w) wtt[w * WTLEN + i] = tanhf(wts[w * WTLEN + i]);
    if (i == 0) {
        float mx = attn[0];
        for (int w = 1; w < NWT; ++w) mx = fmaxf(mx, attn[w]);
        float e[NWT]; float s = 0.f;
        for (int w = 0; w < NWT; ++w) { e[w] = __expf(attn[w] - mx); s += e[w]; }
        for (int w = 0; w < NWT; ++w) ww[w] = e[w] / s;
    }
}

// per-batch exclusive prefix of phase increments over frames
__global__ void k_phasescan(const float* __restrict__ pitch, float* __restrict__ incf,
                            float* __restrict__ pf)
{
    int b = threadIdx.x;
    if (b >= BATCH) return;
    float acc = 0.f;
    for (int t = 0; t < TFR; ++t) {
        float inc = pitch[(size_t)b * TFR + t] * ((float)WTLEN / (float)SRATE);
        incf[b * TFR + t] = inc;
        pf[b * TFR + t]   = acc;
        acc += inc * (float)BLOCK;
    }
}

// wavetable oscillator, adds harmonic into SIG
__global__ void k_harmonic(const float* __restrict__ incf, const float* __restrict__ pf,
                           const float* __restrict__ ta2, const float* __restrict__ wtt,
                           const float* __restrict__ ww, float* __restrict__ sig)
{
    int s = blockIdx.x * blockDim.x + threadIdx.x;
    if (s >= BATCH * ALEN) return;
    int b = s / ALEN, off = s % ALEN;
    int t = off / BLOCK, o = off % BLOCK;
    float inc   = incf[b * TFR + t];
    float phase = pf[b * TFR + t] + (float)o * inc;
    float idx   = fmodf(phase, (float)WTLEN);
    float low   = floorf(idx);
    float alpha = idx - low;
    int li = (int)low;
    int hi = ((int)ceilf(idx)) & (WTLEN - 1);
    float val = 0.f;
#pragma unroll
    for (int w = 0; w < NWT; ++w) {
        float lv = wtt[w * WTLEN + li], hv = wtt[w * WTLEN + hi];
        val += ww[w] * (lv + alpha * (hv - lv));
    }
    sig[(size_t)b * ALEN + off] += val * ta2[b * TFR + t];
}

// reverb impulse: noise * exp-decay * sigmoid(wet), impulse[0]=1
__global__ void k_rvimp(const float* __restrict__ rn, const float* __restrict__ decay,
                        const float* __restrict__ wet, float* __restrict__ imp)
{
    int l = blockIdx.x * blockDim.x + threadIdx.x;
    if (l >= SRATE) return;
    float sp  = log1pf(__expf(-decay[0]));
    float env = __expf(-sp * ((float)l / (float)SRATE) * 500.f);
    float v   = rn[l] * env * sigmoidf_(wet[0]);
    imp[l] = (l == 0) ? 1.f : v;
}

// reverb: y[b][j] = sum_{d<=min(j,15999)} imp[d]*x[b][j-d]  (LDS-tiled taps)
__global__ __launch_bounds__(256)
void k_reverb(const float* __restrict__ sig, const float* __restrict__ imp,
              float* __restrict__ out)
{
    __shared__ float si[256];
    int b = blockIdx.y;
    int j = blockIdx.x * 256 + threadIdx.x;
    const float* x = sig + (size_t)b * ALEN;
    float acc = 0.f;
    for (int d0 = 0; d0 < SRATE; d0 += 256) {
        si[threadIdx.x] = (d0 + (int)threadIdx.x < SRATE) ? imp[d0 + threadIdx.x] : 0.f;
        __syncthreads();
        if (d0 + 256 < SRATE) __builtin_prefetch(&imp[d0 + 256], 0, 1);
        int dmax = min(256, SRATE - d0);
        for (int dd = 0; dd < dmax; ++dd) {
            int d = d0 + dd;
            if (d <= j) acc += si[dd] * x[j - d];
        }
        __syncthreads();
    }
    out[(size_t)b * ALEN + j] = acc;
}

// ---------------------------------------------------------------------------
// host-side launcher
// ---------------------------------------------------------------------------
static inline void gemm(const float* A, int lda, const float* W, int ldw, const float* bias,
                        float* C, int ldc, int M, int N, int K, hipStream_t s)
{
    int tiles = (M / 16) * ((N + 15) / 16);
    int blocks = (tiles + 3) / 4;
    if (N % 16 == 0)
        k_gemm_wmma<false><<<blocks, 128, 0, s>>>(A, lda, W, ldw, bias, C, ldc, M, N, K);
    else
        k_gemm_wmma<true><<<blocks, 128, 0, s>>>(A, lda, W, ldw, bias, C, ldc, M, N, K);
}
static inline void lnleaky(float* X, int ld, int N, const float* g, const float* be,
                           int rows, hipStream_t s)
{
    k_ln_leaky<<<rows, 256, 0, s>>>(X, ld, N, g, be);
}

extern "C" void kernel_launch(void* const* d_in, const int* in_sizes, int n_in,
                              void* d_out, int out_size, void* d_ws, size_t ws_size,
                              hipStream_t stream)
{
    (void)in_sizes; (void)n_in; (void)out_size; (void)ws_size;
    auto F = [&](int i) { return (const float*)d_in[i]; };
    // flatten order (insertion-order DFS of setup_inputs dict):
    // 0 mfcc 1 pitch 2 loudness 3 noise_unit
    // 4 ln.g 5 ln.b | 6 Wih1 7 Whh1 8 bih1 9 bhh1 | 10 W16 11 b16
    // 12..23 in_mlps[0] L0..L2 (W,b,g,beta) | 24..35 in_mlps[1] | 36..47 in_mlps[2]
    // 48 Wih2 49 Whh2 50 bih2 51 bhh2 | 52..63 out_mlp L0..L2
    // 64 lw 65 lb | 66 proj0W 67 proj0b (dead) | 68 proj1W 69 proj1b
    // 70 rnoise 71 decay 72 wet | 73 wavetables 74 attention
    const int ROWS = BATCH * TFR;   // 12800
    const int WHHN = GDIM * HID;    // 786432

    float* ws = (float*)d_ws;
    size_t o = 0;
    auto alloc = [&](size_t n) { float* p = ws + o; o += n; return p; };
    float* XA  = alloc((size_t)ROWS * 20);
    float* BIG = alloc((size_t)ROWS * GDIM);
    float* H1  = alloc((size_t)ROWS * HID);
    float* M16 = alloc((size_t)ROWS * 16);
    float* T0  = alloc((size_t)ROWS * HID);
    float* T1  = alloc((size_t)ROWS * HID);
    float* HC2 = alloc((size_t)ROWS * 2048);
    float* GH  = alloc((size_t)BATCH * GDIM);
    float* P1  = alloc((size_t)ROWS * NBANDS);
    float* IMP = alloc((size_t)ROWS * BLOCK);
    float* SIG = alloc((size_t)BATCH * ALEN);
    float* TA2 = alloc(ROWS);
    float* INC = alloc(ROWS);
    float* PF  = alloc(ROWS);
    float* WTT = alloc(NWT * WTLEN);
    float* WW  = alloc(16);
    float* RIM = alloc(SRATE);
    _Float16* WH16A = (_Float16*)alloc(WHHN / 2);   // Whh1 in f16
    _Float16* WH16B = (_Float16*)alloc(WHHN / 2);   // Whh2 in f16

    // weight pre-conversion for the hot recurrent operand
    k_f2h<<<(WHHN + 255) / 256, 256, 0, stream>>>(F(7),  WH16A, WHHN);
    k_f2h<<<(WHHN + 255) / 256, 256, 0, stream>>>(F(49), WH16B, WHHN);

    // 1) mfcc transpose + LN
    k_mfcc_ln<<<(ROWS + 255) / 256, 256, 0, stream>>>(F(0), F(4), F(5), XA);
    // 2) xs1 = xa @ Wih1^T + bih1
    gemm(XA, 20, F(6), 20, F(8), BIG, GDIM, ROWS, GDIM, 20, stream);
    // 3) GRU #1 (persistent)
    k_gru_seq<<<1, 1024, 0, stream>>>(BIG, WH16A, F(9), GH, H1, HID, TFR);
    // 4) mlp_mfcc 512->16 (bilinear same-size resize == identity)
    gemm(H1, HID, F(10), HID, F(11), M16, 16, ROWS, 16, HID, stream);

    // 5-7) in_mlps -> write into HC2 columns [512..2048)
    const float* ins[3]  = { F(1), F(2), M16 };
    const int    dins[3] = { 1, 1, 16 };
    for (int m = 0; m < 3; ++m) {
        int base = 12 + m * 12;
        float* dst = HC2 + 512 + m * 512;
        gemm(ins[m], dins[m], F(base + 0), dins[m], F(base + 1), T0, HID, ROWS, HID,
             dins[m], stream);
        lnleaky(T0, HID, HID, F(base + 2), F(base + 3), ROWS, stream);
        gemm(T0, HID, F(base + 4), HID, F(base + 5), T1, HID, ROWS, HID, HID, stream);
        lnleaky(T1, HID, HID, F(base + 6), F(base + 7), ROWS, stream);
        gemm(T1, HID, F(base + 8), HID, F(base + 9), dst, 2048, ROWS, HID, HID, stream);
        lnleaky(dst, 2048, HID, F(base + 10), F(base + 11), ROWS, stream);
    }

    // 8) xs2 = hcat @ Wih2^T + bih2  (hcat = HC2 cols 512..2048)
    gemm(HC2 + 512, 2048, F(48), GDIM, F(50), BIG, GDIM, ROWS, GDIM, GDIM, stream);
    // 9) GRU #2 -> HC2 cols [0..512)
    k_gru_seq<<<1, 1024, 0, stream>>>(BIG, WH16B, F(51), GH, HC2, 2048, TFR);

    // 10) out_mlp over HC2 (2048-wide)
    gemm(HC2, 2048, F(52), 2048, F(53), T0, HID, ROWS, HID, 2048, stream);
    lnleaky(T0, HID, HID, F(54), F(55), ROWS, stream);
    gemm(T0, HID, F(56), HID, F(57), T1, HID, ROWS, HID, HID, stream);
    lnleaky(T1, HID, HID, F(58), F(59), ROWS, stream);
    gemm(T1, HID, F(60), HID, F(61), T0, HID, ROWS, HID, HID, stream);
    lnleaky(T0, HID, HID, F(62), F(63), ROWS, stream);

    // 11) p1 = h @ proj1^T + b, then scale_function(p1 - 5)
    gemm(T0, HID, F(68), HID, F(69), P1, NBANDS, ROWS, NBANDS, HID, stream);
    k_scale<<<(ROWS * NBANDS + 255) / 256, 256, 0, stream>>>(P1, ROWS * NBANDS);

    // 12) impulse responses + per-frame noise filtering -> SIG
    k_impulse<<<ROWS, BLOCK, 0, stream>>>(P1, IMP);
    k_noiseconv<<<ROWS, BLOCK, 0, stream>>>(F(3), IMP, SIG);

    // 13) wavetable harmonic, added into SIG
    k_ta2<<<(ROWS + 255) / 256, 256, 0, stream>>>(F(2), F(64), F(65), TA2, ROWS);
    k_wtprep<<<1, WTLEN, 0, stream>>>(F(73), F(74), WTT, WW);
    k_phasescan<<<1, 32, 0, stream>>>(F(1), INC, PF);
    k_harmonic<<<(BATCH * ALEN + 255) / 256, 256, 0, stream>>>(INC, PF, TA2, WTT, WW, SIG);

    // 14) reverb -> d_out
    k_rvimp<<<(SRATE + 255) / 256, 256, 0, stream>>>(F(70), F(71), F(72), RIM);
    dim3 rg(ALEN / 256, BATCH);
    k_reverb<<<rg, 256, 0, stream>>>(SIG, RIM, (float*)d_out);
}
